// RecurrentLayer_7232724926962
// MI455X (gfx1250) — matrix operand order repaired and checked by reading, not compiled, not run on hardware
//
#include <hip/hip_runtime.h>
#include <hip/hip_bf16.h>

typedef __attribute__((ext_vector_type(16))) __bf16 v16bf;
typedef __attribute__((ext_vector_type(8)))  float  v8f;

#define T_STEPS 1024
#define BATCH   64
#define DIN     256
#define CELL    128
#define GATES   512

__device__ __forceinline__ float sigmoidf_(float x) {
    return 1.0f / (1.0f + __expf(-x));
}
__device__ __forceinline__ float tanhf_(float x) {
    x = fminf(fmaxf(x, -15.0f), 15.0f);
    float e = __expf(2.0f * x);
    return (e - 1.0f) / (e + 1.0f);
}

// ---------------------------------------------------------------------------
// Exclusive prefix sum of batch_sizes -> offsets (Hillis-Steele in LDS)
// ---------------------------------------------------------------------------
__global__ __launch_bounds__(1024)
void offsets_kernel(const int* __restrict__ bs, int* __restrict__ offs) {
    __shared__ int s[T_STEPS];
    int t = threadIdx.x;
    int my = bs[t];
    s[t] = my;
    __syncthreads();
    for (int d = 1; d < T_STEPS; d <<= 1) {
        int v = (t >= d) ? s[t - d] : 0;
        __syncthreads();
        s[t] += v;
        __syncthreads();
    }
    offs[t] = s[t] - my;
}

// ---------------------------------------------------------------------------
// Gin[dir][m][g] = x[m][:] @ W_ih[k]^T + b_ih[k] + b_hh[k]
// M = T*B = 65536 (padded), N = 512, K = 256, bf16 WMMA, f32 accumulate.
// layer 0: x gathered from packed input via offsets (branchless: row clamped
//          to 0 and scaled by 0 when b >= bs[t])
// layer 1: x = x1 (bf16, written by layer-0 recurrence)
// grid (512, 8, 2) x 256 threads. Tile: 128(M) x 64(N), k-chunks of 64.
// ---------------------------------------------------------------------------
__global__ __launch_bounds__(256)
void gates_gemm_kernel(const float* __restrict__ input,
                       const __bf16* __restrict__ x1,
                       const float* __restrict__ W_ih,
                       const float* __restrict__ b_ih,
                       const float* __restrict__ b_hh,
                       const int* __restrict__ bs,
                       const int* __restrict__ offs,
                       float* __restrict__ Gin0,
                       float* __restrict__ Gin1,
                       int layer)
{
    const int dir = blockIdx.z;
    const int k   = 2 * layer + dir;
    float* __restrict__ Gin = dir ? Gin1 : Gin0;
    const float* __restrict__ Wk = W_ih + (size_t)k * GATES * DIN;

    __shared__ __bf16 As[128][64];   // 16 KB
    __shared__ __bf16 Bs[64][64];    // 8 KB

    const int tid  = threadIdx.x;
    const int lane = tid & 31;
    const int w    = tid >> 5;                  // 8 waves
    const int mbase = blockIdx.x * 128;
    const int nbase = blockIdx.y * 64;
    const int wm = (w & 3) * 2;                 // wave's M-tile pair
    const int wn = (w >> 2) * 2;                // wave's N-tile pair

    // branchless gather setup for this thread's staged A row
    const int arow  = tid & 127;                // LDS row
    const int acol  = (tid >> 7) * 32;          // 32-col half
    float  ascale = 1.0f;
    size_t asrc_row;
    if (layer == 0) {
        int m  = mbase + arow;
        int tt = m >> 6, bb = m & 63;
        bool valid = bb < bs[tt];
        ascale   = valid ? 1.0f : 0.0f;
        asrc_row = valid ? (size_t)(offs[tt] + bb) : 0;   // clamp: row 0 is valid
    } else {
        asrc_row = (size_t)(mbase + arow);
    }

    v8f C[2][2] = {};

    for (int kc = 0; kc < DIN; kc += 64) {
        // ---- stage A chunk (128 x 64 bf16), branchless ----
        {
            __bf16* dst = &As[arow][acol];
            if (layer == 0) {
                const float* src = input + asrc_row * DIN + kc + acol;
                #pragma unroll
                for (int i = 0; i < 32; i++) dst[i] = (__bf16)(src[i] * ascale);
            } else {
                const __bf16* src = x1 + asrc_row * DIN + kc + acol;
                #pragma unroll
                for (int i = 0; i < 32; i++) dst[i] = src[i];
            }
        }
        // ---- stage B chunk (64 x 64 bf16) from W_ih (f32 -> bf16) ----
        {
            int row = tid >> 2;                 // 0..63
            int seg = (tid & 3) * 16;
            const float* src = Wk + (size_t)(nbase + row) * DIN + kc + seg;
            #pragma unroll
            for (int i = 0; i < 16; i++) Bs[row][seg + i] = (__bf16)src[i];
        }
        __syncthreads();

        // ---- fragments + WMMA: two k-steps of 32 per staged chunk ----
        #pragma unroll
        for (int ks = 0; ks < 2; ks++) {
            v16bf Bf[2];
            #pragma unroll
            for (int ni = 0; ni < 2; ni++) {
                int n = (wn + ni) * 16 + (lane & 15);
                const __bf16* p = &Bs[n][ks * 32 + (lane >> 4) * 8]; // two 16B runs (k, k+16)
                #pragma unroll
                for (int i = 0; i < 8; i++) { Bf[ni][i] = p[i]; Bf[ni][8 + i] = p[16 + i]; }
            }
            #pragma unroll
            for (int mi = 0; mi < 2; mi++) {
                int m  = (wm + mi) * 16 + (lane & 15);
                int kb = ks * 32 + (lane >> 4) * 8;
                v16bf Af;
                const __bf16* pl = &As[m][kb];
                const __bf16* ph = &As[m][kb + 16];
                #pragma unroll
                for (int i = 0; i < 8; i++) { Af[i] = pl[i]; Af[8 + i] = ph[i]; }
                #pragma unroll
                for (int ni = 0; ni < 2; ni++)
                    C[mi][ni] = __builtin_amdgcn_wmma_f32_16x16x32_bf16(
                        false, Af, false, Bf[ni], (short)0, C[mi][ni],
                        false, false);
            }
        }
        __syncthreads();
    }

    // ---- epilogue: bias add + store ----
    #pragma unroll
    for (int mi = 0; mi < 2; mi++) {
        #pragma unroll
        for (int ni = 0; ni < 2; ni++) {
            int ncol  = nbase + (wn + ni) * 16 + (lane & 15);
            float bia = b_ih[k * GATES + ncol] + b_hh[k * GATES + ncol];
            int mrow0 = mbase + (wm + mi) * 16 + (lane >> 4) * 8;
            #pragma unroll
            for (int r = 0; r < 8; r++)
                Gin[(size_t)(mrow0 + r) * GATES + ncol] = C[mi][ni][r] + bia;
        }
    }
}

// ---------------------------------------------------------------------------
// Sequential LSTM scan. grid (4 batch-groups of 16 rows, 2 dirs) x 512 thr.
// Per step: gates(16x512) = h(16x128)@Whh^T via bf16 WMMA (Whh fragments
// register-resident, time invariant) + Gin[t]; then masked cell update.
// layer 0 writes x1 (bf16 padded, concat fwd/bwd); layer 1 writes packed
// d_out directly. Both write h_n/c_n for k = 2*layer+dir.
// ---------------------------------------------------------------------------
__global__ __launch_bounds__(512)
void lstm_scan_kernel(const float* __restrict__ Gin0,
                      const float* __restrict__ Gin1,
                      const float* __restrict__ W_hh,
                      const float* __restrict__ h0,
                      const float* __restrict__ c0,
                      const int* __restrict__ bs,
                      const int* __restrict__ offs,
                      __bf16* __restrict__ x1,
                      float* __restrict__ out,
                      int layer, int total)
{
    const int grp = blockIdx.x;          // 0..3 : batch rows [16*grp, 16*grp+16)
    const int dir = blockIdx.y;          // 0 fwd, 1 bwd
    const int k   = 2 * layer + dir;
    const float* __restrict__ Gin = dir ? Gin1 : Gin0;
    const int bg0 = grp * 16;

    __shared__ float  gates[16][GATES];  // 32 KB: h@Whh^T per step
    __shared__ __bf16 hbuf[16][CELL];    // 4 KB: h carry (bf16 for WMMA A)

    const int tid  = threadIdx.x;
    const int lane = tid & 31;
    const int w    = tid >> 5;           // 16 waves; wave w owns N-tiles 2w,2w+1

    // elementwise mapping: row bl, 4 cells at j0
    const int bl = tid >> 5;             // 0..15
    const int j0 = (tid & 31) * 4;

    // ---- init h,c in registers + h bf16 buffer ----
    float hreg[4], creg[4];
    {
        const float* hp = h0 + ((size_t)k * BATCH + bg0 + bl) * CELL + j0;
        const float* cp = c0 + ((size_t)k * BATCH + bg0 + bl) * CELL + j0;
        #pragma unroll
        for (int i = 0; i < 4; i++) { hreg[i] = hp[i]; creg[i] = cp[i]; }
        #pragma unroll
        for (int i = 0; i < 4; i++) hbuf[bl][j0 + i] = (__bf16)hreg[i];
    }

    // ---- hoist Whh B-fragments to registers (time invariant) ----
    // B(k_dim, n) = Whh[n][k_dim]; Whh is row-major (512 x 128) f32.
    v16bf Bf[4][2];
    {
        const float* Wk = W_hh + (size_t)k * GATES * CELL;
        #pragma unroll
        for (int ks = 0; ks < 4; ks++) {
            #pragma unroll
            for (int ni = 0; ni < 2; ni++) {
                int n = (2 * w + ni) * 16 + (lane & 15);
                const float* p = Wk + (size_t)n * CELL + ks * 32 + (lane >> 4) * 8;
                #pragma unroll
                for (int i = 0; i < 8; i++) { Bf[ks][ni][i] = (__bf16)p[i]; Bf[ks][ni][8 + i] = (__bf16)p[16 + i]; }
            }
        }
    }

    const size_t HN = (size_t)total * 256;        // h_n base in out
    const size_t CN = HN + (size_t)4 * BATCH * CELL;

    for (int step = 0; step < T_STEPS; step++) {
        const int t = dir ? (T_STEPS - 1 - step) : step;
        __syncthreads();   // h carry (hbuf) visible to all waves

        // ---- h(16x128) @ Whh^T -> this wave's 16 x 32 gate columns ----
        v8f Cn[2] = {};
        #pragma unroll
        for (int ks = 0; ks < 4; ks++) {
            v16bf Af;
            {
                int m  = lane & 15;
                int kb = ks * 32 + (lane >> 4) * 8;
                const __bf16* pl = &hbuf[m][kb];
                #pragma unroll
                for (int i = 0; i < 8; i++) { Af[i] = pl[i]; Af[8 + i] = pl[16 + i]; }
            }
            #pragma unroll
            for (int ni = 0; ni < 2; ni++)
                Cn[ni] = __builtin_amdgcn_wmma_f32_16x16x32_bf16(
                    false, Af, false, Bf[ks][ni], (short)0, Cn[ni], false, false);
        }
        #pragma unroll
        for (int ni = 0; ni < 2; ni++) {
            int ncol = (2 * w + ni) * 16 + (lane & 15);
            int m0   = (lane >> 4) * 8;
            #pragma unroll
            for (int r = 0; r < 8; r++) gates[m0 + r][ncol] = Cn[ni][r];
        }
        __syncthreads();   // gates ready (and hbuf reads retired)

        // ---- elementwise cell update ----
        const int bglob = bg0 + bl;
        const int bs_t  = bs[t];
        const size_t grow = ((size_t)t * BATCH + bglob) * GATES;
        float gi[4], gf[4], gg[4], go[4];
        #pragma unroll
        for (int i = 0; i < 4; i++) {
            gi[i] = gates[bl][j0 + i]            + Gin[grow + j0 + i];
            gf[i] = gates[bl][CELL + j0 + i]     + Gin[grow + CELL + j0 + i];
            gg[i] = gates[bl][2 * CELL + j0 + i] + Gin[grow + 2 * CELL + j0 + i];
            go[i] = gates[bl][3 * CELL + j0 + i] + Gin[grow + 3 * CELL + j0 + i];
        }
        // prefetch next step's Gin row into cache
        if (step + 1 < T_STEPS) {
            const int tn = dir ? (t - 1) : (t + 1);
            __builtin_prefetch(&Gin[((size_t)tn * BATCH + bglob) * GATES + j0], 0, 0);
        }
        const bool m = bglob < bs_t;
        #pragma unroll
        for (int i = 0; i < 4; i++) {
            float c2 = sigmoidf_(gf[i]) * creg[i] + sigmoidf_(gi[i]) * tanhf_(gg[i]);
            float h2 = sigmoidf_(go[i]) * tanhf_(c2);
            if (m) { creg[i] = c2; hreg[i] = h2; }
        }
        #pragma unroll
        for (int i = 0; i < 4; i++) hbuf[bl][j0 + i] = (__bf16)hreg[i];

        if (layer == 0) {
            __bf16* xp = x1 + ((size_t)t * BATCH + bglob) * 256 + dir * CELL + j0;
            #pragma unroll
            for (int i = 0; i < 4; i++) xp[i] = (__bf16)hreg[i];
        } else if (m) {
            float* op = out + ((size_t)(offs[t] + bglob)) * 256 + dir * CELL + j0;
            #pragma unroll
            for (int i = 0; i < 4; i++) op[i] = hreg[i];
        }
    }

    // ---- final h_n / c_n ----
    {
        float* hp = out + HN + ((size_t)k * BATCH + bg0 + bl) * CELL + j0;
        float* cp = out + CN + ((size_t)k * BATCH + bg0 + bl) * CELL + j0;
        #pragma unroll
        for (int i = 0; i < 4; i++) { hp[i] = hreg[i]; cp[i] = creg[i]; }
    }
}

// ---------------------------------------------------------------------------
extern "C" void kernel_launch(void* const* d_in, const int* in_sizes, int n_in,
                              void* d_out, int out_size, void* d_ws, size_t ws_size,
                              hipStream_t stream)
{
    const float* input = (const float*)d_in[0];
    const int*   bsz   = (const int*)  d_in[1];
    const float* h0    = (const float*)d_in[2];
    const float* c0    = (const float*)d_in[3];
    const float* W_ih  = (const float*)d_in[4];
    const float* W_hh  = (const float*)d_in[5];
    const float* b_ih  = (const float*)d_in[6];
    const float* b_hh  = (const float*)d_in[7];
    float* out = (float*)d_out;
    const int total = in_sizes[0] / DIN;

    // workspace layout: offsets(4KB) | Gin0(128MB) | Gin1(128MB) | x1(32MB)
    char* w = (char*)d_ws;
    int*    offs = (int*)w;
    float*  Gin0 = (float*)(w + 4096);
    float*  Gin1 = Gin0 + (size_t)T_STEPS * BATCH * GATES;
    __bf16* x1   = (__bf16*)(Gin1 + (size_t)T_STEPS * BATCH * GATES);
    (void)ws_size; (void)n_in; (void)out_size;

    offsets_kernel<<<1, 1024, 0, stream>>>(bsz, offs);

    dim3 ggrid(T_STEPS * BATCH / 128, GATES / 64, 2);   // 512 x 8 x 2
    dim3 sgrid(BATCH / 16, 2);                          // 4 x 2
    for (int layer = 0; layer < 2; layer++) {
        gates_gemm_kernel<<<ggrid, 256, 0, stream>>>(
            input, x1, W_ih, b_ih, b_hh, bsz, offs, Gin0, Gin1, layer);
        lstm_scan_kernel<<<sgrid, 512, 0, stream>>>(
            Gin0, Gin1, W_hh, h0, c0, bsz, offs, x1, out, layer, total);
    }
}
